// Conditioner_80178449482393
// MI455X (gfx1250) — matrix operand order repaired and checked by reading, not compiled, not run on hardware
//
#include <hip/hip_runtime.h>
#include <math.h>

#define D_MODEL 1408
#define HEADS 16
#define DHEAD 88
#define TKV 1088
#define BATCH 4
#define FRAMES 16

typedef __attribute__((ext_vector_type(16))) __bf16 v16bf;
typedef __attribute__((ext_vector_type(8)))  float v8f;
typedef __attribute__((ext_vector_type(8)))  unsigned short ushort8;

union FragBF { v16bf v; ushort8 h[2]; };

__device__ inline unsigned short f32_to_bf16(float f) {
  union { float f; unsigned int u; } cv; cv.f = f;
  unsigned int u = cv.u;
  u += 0x7FFFu + ((u >> 16) & 1u);       // round to nearest even
  return (unsigned short)(u >> 16);
}

// ---------------------------------------------------------------------------
// Tiled GEMM: C[M,N] = act(A[M,K] @ B[K,N] + bias) + resid
// f32 operands converted to bf16 in LDS, v_wmma_f32_16x16x32_bf16 accumulate.
// BM=64, BN=128, BK=64; 256 threads = 8 waves in 4x2; each wave owns a 16x64
// C strip = 4 accumulators; 8 WMMAs per wave per K-tile (one A fragment is
// reused across 4 B fragments per 32-wide K subtile).
// Requires M%64==0, N%128==0, K%64==0 (true for every GEMM in this model).
// ---------------------------------------------------------------------------
#define BM 64
#define BN 128
#define BK 64

__global__ __launch_bounds__(256) void gemm_bf16_kernel(
    const float* __restrict__ A, const float* __restrict__ B, float* __restrict__ C,
    const float* __restrict__ bias, const float* __restrict__ resid,
    int M, int N, int K, int act)
{
  __shared__ unsigned short As[BM * BK];   // row-major [m][k]   (8 KB)
  __shared__ unsigned short Bs[BN * BK];   // transposed [n][k] (16 KB)
  const int tid  = threadIdx.x;
  const int lane = tid & 31;
  const int wave = tid >> 5;
  const int wr = wave & 3;         // 0..3 -> row group of 16
  const int wc = wave >> 2;        // 0..1 -> col group of 64
  const int lm = lane & 15;
  const int g  = lane >> 4;        // lane half
  const int m0 = blockIdx.y * BM;
  const int n0 = blockIdx.x * BN;

  v8f acc[4] = {};

  for (int k0 = 0; k0 < K; k0 += BK) {
#pragma unroll
    for (int i = 0; i < 16; ++i) {          // A tile 64x64
      int idx = tid + i * 256;
      int m = idx >> 6, k = idx & 63;
      As[idx] = f32_to_bf16(A[(size_t)(m0 + m) * K + k0 + k]);
    }
#pragma unroll
    for (int i = 0; i < 32; ++i) {          // B tile 64x128 -> transposed
      int idx = tid + i * 256;
      int k = idx >> 7, n = idx & 127;
      Bs[n * BK + k] = f32_to_bf16(B[(size_t)(k0 + k) * N + n0 + n]);
    }
    // Prefetch next K-tile (global_prefetch_b8) while we compute this one.
    if (k0 + BK < K) {
      if (tid < BM) __builtin_prefetch(&A[(size_t)(m0 + tid) * K + k0 + BK], 0, 1);
      if (tid < BK) __builtin_prefetch(&B[(size_t)(k0 + BK + tid) * N + n0], 0, 1);
    }
    __syncthreads();

#pragma unroll
    for (int ks = 0; ks < BK; ks += 32) {
      // A fragment (16x32, ISA layout): lane half g=0 holds K {0..7,16..23},
      // g=1 holds K {8..15,24..31}, row = lm.
      FragBF fa;
      const unsigned short* arow = &As[(wr * 16 + lm) * BK + ks];
      fa.h[0] = *(const ushort8*)(arow + 8 * g);
      fa.h[1] = *(const ushort8*)(arow + 16 + 8 * g);
#pragma unroll
      for (int f = 0; f < 4; ++f) {
        // B fragment (32x16): lane column fixed, lane half g holds K {8g..8g+7, 16+8g..16+8g+7}.
        FragBF fb;
        const unsigned short* brow = &Bs[(wc * 64 + f * 16 + lm) * BK + ks + 8 * g];
        fb.h[0] = ((const ushort8*)brow)[0];
        fb.h[1] = ((const ushort8*)brow)[2];
        acc[f] = __builtin_amdgcn_wmma_f32_16x16x32_bf16(
            false, fa.v, false, fb.v, (short)0, acc[f], false, false);
      }
    }
    __syncthreads();
  }

#pragma unroll
  for (int f = 0; f < 4; ++f) {
    int n = n0 + wc * 64 + f * 16 + lm;
#pragma unroll
    for (int r = 0; r < 8; ++r) {
      int m = m0 + wr * 16 + 8 * g + r;     // C/D layout: VGPR r -> M = r + 8*half
      float val = acc[f][r];
      if (bias) val += bias[n];
      if (act == 1) val = 0.5f * val * (1.f + erff(val * 0.70710678118654752f));
      if (resid) val += resid[(size_t)m * N + n];
      C[(size_t)m * N + n] = val;
    }
  }
}

// ---------------------------------------------------------------------------
// Row LayerNorm over D_MODEL with output-row remapping:
//   out_row = (row / rpb_in) * out_stride_rows + out_row_off + (row % rpb_in)
// Used to write LN(xe) and LN(lat) directly into the concatenated kv buffer.
// ---------------------------------------------------------------------------
__device__ inline float block_sum_256(float v) {
  __shared__ float red[8];
  for (int off = 16; off; off >>= 1) v += __shfl_xor(v, off, 32);
  int lane = threadIdx.x & 31, wv = threadIdx.x >> 5;
  if (lane == 0) red[wv] = v;
  __syncthreads();
  if (wv == 0) {
    float t = (lane < 8) ? red[lane] : 0.f;
    for (int off = 4; off; off >>= 1) t += __shfl_xor(t, off, 32);
    if (lane == 0) red[0] = t;
  }
  __syncthreads();
  v = red[0];
  __syncthreads();
  return v;
}

__global__ __launch_bounds__(256) void layernorm_kernel(
    const float* __restrict__ in, float* __restrict__ out,
    const float* __restrict__ gam, const float* __restrict__ bet,
    int rpb_in, int out_stride_rows, int out_row_off)
{
  int row = blockIdx.x;
  int orow = (row / rpb_in) * out_stride_rows + out_row_off + (row % rpb_in);
  const float* x = in + (size_t)row * D_MODEL;
  float* y = out + (size_t)orow * D_MODEL;
  float s = 0.f;
  for (int i = threadIdx.x; i < D_MODEL; i += 256) s += x[i];
  s = block_sum_256(s);
  float mu = s * (1.f / D_MODEL);
  float v = 0.f;
  for (int i = threadIdx.x; i < D_MODEL; i += 256) { float d = x[i] - mu; v += d * d; }
  v = block_sum_256(v);
  float rstd = rsqrtf(v * (1.f / D_MODEL) + 1e-5f);
  for (int i = threadIdx.x; i < D_MODEL; i += 256)
    y[i] = (x[i] - mu) * rstd * gam[i] + bet[i];
}

// ---------------------------------------------------------------------------
// Attention: one block per (head, batch); 4 waves, each wave = one query at a
// time with scores in LDS, masked softmax, weighted V sum. DH=88.
// ---------------------------------------------------------------------------
__global__ __launch_bounds__(128) void attention_kernel(
    const float* __restrict__ q, const float* __restrict__ kv,
    const int* __restrict__ idxs, float* __restrict__ out)
{
  const int h = blockIdx.x, b = blockIdx.y;
  const int wave = threadIdx.x >> 5, lane = threadIdx.x & 31;
  __shared__ float sc[4][TKV];
  __shared__ float qs[4][DHEAD];
  const float scale = 0.10660035817780522f;   // 1/sqrt(88)
  const int maskStart = idxs[b] * 64;

  for (int qi = wave; qi < 64; qi += 4) {
    const float* qrow = q + ((size_t)(b * 64 + qi)) * D_MODEL + h * DHEAD;
    for (int d = lane; d < DHEAD; d += 32) qs[wave][d] = qrow[d];
    __syncthreads();

    float mx = -INFINITY;
    for (int j = lane; j < TKV; j += 32) {
      const float* krow = kv + ((size_t)(b * TKV + j)) * (2 * D_MODEL) + h * DHEAD;
      float s = 0.f;
      for (int d = 0; d < DHEAD; ++d) s += qs[wave][d] * krow[d];
      s *= scale;
      if (j >= maskStart && j < TKV - 64) s = -INFINITY;
      sc[wave][j] = s;
      mx = fmaxf(mx, s);
    }
    for (int off = 16; off; off >>= 1) mx = fmaxf(mx, __shfl_xor(mx, off, 32));
    float ls = 0.f;
    for (int j = lane; j < TKV; j += 32) {
      float p = expf(sc[wave][j] - mx);
      sc[wave][j] = p; ls += p;
    }
    for (int off = 16; off; off >>= 1) ls += __shfl_xor(ls, off, 32);
    float inv = 1.f / ls;
    __syncthreads();

    float* orow = out + ((size_t)(b * 64 + qi)) * D_MODEL + h * DHEAD;
    for (int d = lane; d < DHEAD; d += 32) {
      const float* vcol = kv + (size_t)(b * TKV) * (2 * D_MODEL) + D_MODEL + h * DHEAD + d;
      float acc = 0.f;
      for (int j = 0; j < TKV; ++j) acc += sc[wave][j] * vcol[(size_t)j * (2 * D_MODEL)];
      orow[d] = acc * inv;
    }
    __syncthreads();
  }
}

// ---------------------------------------------------------------------------
// Small setup kernels
// ---------------------------------------------------------------------------
__global__ void pos_embed_kernel(float* pos) {        // [64, 1408] sincos-2d
  int idx = blockIdx.x * 256 + threadIdx.x;
  if (idx >= 64 * D_MODEL) return;
  int m = idx / D_MODEL, col = idx % D_MODEL;
  int gi = m >> 3, gj = m & 7;
  int seg = col / 352, t = col - seg * 352;
  float om = powf(10000.f, -(float)t / 352.f);
  float p = (float)((seg < 2) ? gj : gi);
  float a = p * om;
  pos[idx] = (seg & 1) ? cosf(a) : sinf(a);
}

__global__ void transpose_kernel(const float* in, float* out, int R, int C) {
  size_t idx = (size_t)blockIdx.x * 256 + threadIdx.x;
  if (idx >= (size_t)R * C) return;
  int r = (int)(idx / C), c = (int)(idx % C);
  out[(size_t)c * R + r] = in[idx];
}

__global__ void im2col_kernel(const float* __restrict__ x, float* __restrict__ A) {
  size_t idx = (size_t)blockIdx.x * 256 + threadIdx.x;   // 4096*256 rows*K
  if (idx >= (size_t)4096 * 256) return;
  int col = (int)(idx & 255), row = (int)(idx >> 8);
  int c = col >> 6, p = (col >> 3) & 7, qq = col & 7;
  int pi = row & 63, bf = row >> 6;
  int i = pi >> 3, j = pi & 7;
  A[idx] = x[(((size_t)bf * 4 + c) * 64 + i * 8 + p) * 64 + j * 8 + qq];
}

__global__ void add_pos_media_kernel(float* __restrict__ xe,
                                     const float* __restrict__ pos,
                                     const float* __restrict__ media) {
  size_t idx = (size_t)blockIdx.x * 256 + threadIdx.x;
  if (idx >= (size_t)4096 * D_MODEL) return;
  int n = (int)(idx % D_MODEL);
  int row = (int)(idx / D_MODEL);
  int pi = row & 63, f = (row >> 6) & 15;
  xe[idx] += pos[pi * D_MODEL + n] + media[f * D_MODEL + n];
}

__global__ void broadcast_latents_kernel(const float* __restrict__ lat_in,
                                         float* __restrict__ lat) {
  int idx = blockIdx.x * 256 + threadIdx.x;
  if (idx >= 256 * D_MODEL) return;
  int n = idx % D_MODEL, r = (idx / D_MODEL) & 63;
  lat[idx] = lat_in[r * D_MODEL + n];
}

// ---------------------------------------------------------------------------
extern "C" void kernel_launch(void* const* d_in, const int* in_sizes, int n_in,
                              void* d_out, int out_size, void* d_ws, size_t ws_size,
                              hipStream_t stream) {
  (void)n_in; (void)out_size; (void)ws_size;
  const float *x, *patch_w, *patch_b, *latents, *media_pos, *norm_g, *norm_b;
  const int* idxs;
  const float *nm_g[6], *nm_b[6], *nl_g[6], *nl_b[6], *q_w[6], *q_b[6],
              *kv_w[6], *kv_b[6], *o_w[6], *o_b[6], *ff_g[6], *ff_b[6],
              *w1[6], *w2[6];
  auto F = [&](int i) { return (const float*)d_in[i]; };

  if (in_sizes[0] == BATCH * FRAMES * 4 * 64 * 64) {
    // insertion-order flattening: x, idxs, then params in definition order
    x = F(0); idxs = (const int*)d_in[1];
    patch_w = F(2); patch_b = F(3); latents = F(4); media_pos = F(5);
    norm_g = F(6); norm_b = F(7);
    for (int l = 0; l < 6; ++l) {
      int p = 8 + l * 14;
      nm_g[l] = F(p + 0); nm_b[l] = F(p + 1); nl_g[l] = F(p + 2); nl_b[l] = F(p + 3);
      q_w[l] = F(p + 4);  q_b[l] = F(p + 5);  kv_w[l] = F(p + 6); kv_b[l] = F(p + 7);
      o_w[l] = F(p + 8);  o_b[l] = F(p + 9);  ff_g[l] = F(p + 10); ff_b[l] = F(p + 11);
      w1[l] = F(p + 12);  w2[l] = F(p + 13);
    }
  } else {
    // jax pytree flattening (dict keys sorted): params..., idxs, x
    latents = F(0);
    for (int l = 0; l < 6; ++l) {
      int p = 1 + l * 14;
      kv_b[l] = F(p + 0); kv_w[l] = F(p + 1); nl_b[l] = F(p + 2); nl_g[l] = F(p + 3);
      nm_b[l] = F(p + 4); nm_g[l] = F(p + 5); o_b[l] = F(p + 6);  o_w[l] = F(p + 7);
      q_b[l] = F(p + 8);  q_w[l] = F(p + 9);  ff_b[l] = F(p + 10); ff_g[l] = F(p + 11);
      w1[l] = F(p + 12);  w2[l] = F(p + 13);
    }
    media_pos = F(85); norm_b = F(86); norm_g = F(87); patch_b = F(88); patch_w = F(89);
    idxs = (const int*)d_in[90]; x = F(91);
  }

  float* w = (float*)d_ws;
  size_t o = 0;
  float* POS  = w + o; o += (size_t)64 * D_MODEL;
  float* PWT  = w + o; o += (size_t)256 * D_MODEL;
  float* APE  = w + o; o += (size_t)4096 * 256;
  float* XE   = w + o; o += (size_t)4096 * D_MODEL;
  float* KVIN = w + o; o += (size_t)4352 * D_MODEL;
  float* LNL  = w + o; o += (size_t)256 * D_MODEL;
  float* Qb   = w + o; o += (size_t)256 * D_MODEL;
  float* KVb  = w + o; o += (size_t)4352 * 2816;
  float* AO   = w + o; o += (size_t)256 * D_MODEL;
  float* LAT  = w + o; o += (size_t)256 * D_MODEL;
  float* FLN  = w + o; o += (size_t)256 * D_MODEL;
  float* FFH  = w + o; o += (size_t)256 * 5632;

  // Patch embedding path
  pos_embed_kernel<<<(64 * D_MODEL + 255) / 256, 256, 0, stream>>>(POS);
  transpose_kernel<<<(D_MODEL * 256 + 255) / 256, 256, 0, stream>>>(patch_w, PWT, D_MODEL, 256);
  im2col_kernel<<<4096, 256, 0, stream>>>(x, APE);
  gemm_bf16_kernel<<<dim3(D_MODEL / BN, 4096 / BM), 256, 0, stream>>>(
      APE, PWT, XE, patch_b, nullptr, 4096, D_MODEL, 256, 0);
  add_pos_media_kernel<<<(unsigned)(((size_t)4096 * D_MODEL + 255) / 256), 256, 0, stream>>>(
      XE, POS, media_pos);
  broadcast_latents_kernel<<<(256 * D_MODEL + 255) / 256, 256, 0, stream>>>(latents, LAT);

  for (int l = 0; l < 6; ++l) {
    // LN(xe) -> kv_in[:, 0:1024, :], LN(lat) -> lnlat and kv_in[:, 1024:1088, :]
    layernorm_kernel<<<4096, 256, 0, stream>>>(XE, KVIN, nm_g[l], nm_b[l], 1024, 1088, 0);
    layernorm_kernel<<<256, 256, 0, stream>>>(LAT, LNL, nl_g[l], nl_b[l], 64, 64, 0);
    layernorm_kernel<<<256, 256, 0, stream>>>(LAT, KVIN, nl_g[l], nl_b[l], 64, 1088, 1024);
    // projections
    gemm_bf16_kernel<<<dim3(D_MODEL / BN, 256 / BM), 256, 0, stream>>>(
        LNL, q_w[l], Qb, q_b[l], nullptr, 256, D_MODEL, D_MODEL, 0);
    gemm_bf16_kernel<<<dim3(2816 / BN, 4352 / BM), 256, 0, stream>>>(
        KVIN, kv_w[l], KVb, kv_b[l], nullptr, 4352, 2816, D_MODEL, 0);
    // attention + output projection with residual
    attention_kernel<<<dim3(HEADS, BATCH), 128, 0, stream>>>(Qb, KVb, idxs, AO);
    gemm_bf16_kernel<<<dim3(D_MODEL / BN, 256 / BM), 256, 0, stream>>>(
        AO, o_w[l], LAT, o_b[l], LAT, 256, D_MODEL, D_MODEL, 0);
    // MLP with exact GELU, residual
    layernorm_kernel<<<256, 256, 0, stream>>>(LAT, FLN, ff_g[l], ff_b[l], 64, 64, 0);
    gemm_bf16_kernel<<<dim3(5632 / BN, 256 / BM), 256, 0, stream>>>(
        FLN, w1[l], FFH, nullptr, nullptr, 256, 5632, D_MODEL, 1);
    gemm_bf16_kernel<<<dim3(D_MODEL / BN, 256 / BM), 256, 0, stream>>>(
        FFH, w2[l], LAT, nullptr, LAT, 256, D_MODEL, 5632, 0);
  }
  layernorm_kernel<<<256, 256, 0, stream>>>(LAT, (float*)d_out, norm_g, norm_b, 64, 64, 0);
}
